// TopNets_27968827031916
// MI455X (gfx1250) — compile-verified, matrix operand order checked by reading, not run on hardware
//
#include <hip/hip_runtime.h>
#include <hip/hip_bf16.h>
#include <math.h>

#define NN 20000
#define EE 320000
#define NB 32

typedef __attribute__((ext_vector_type(16))) __bf16 v16bf;
typedef __attribute__((ext_vector_type(8)))  float  v8f;

#define ACT_NONE 0
#define ACT_RELU 1
#define ACT_LEAKY 2

// ---------------- order-preserving float<->uint for atomic max ----------------
__device__ __forceinline__ unsigned fenc(float f) {
    unsigned u = __float_as_uint(f);
    return (u & 0x80000000u) ? ~u : (u | 0x80000000u);
}
__device__ __forceinline__ float fdec(unsigned u) {
    unsigned b = (u & 0x80000000u) ? (u & 0x7FFFFFFFu) : ~u;
    return __uint_as_float(b);
}

__device__ __forceinline__ float act_apply(float v, int act) {
    if (act == ACT_RELU)  return fmaxf(v, 0.f);
    if (act == ACT_LEAKY) return v > 0.f ? v : 0.3f * v;
    return v;
}

// ---------------- weight pre-pack into WMMA B-fragment layout -----------------
// Wp is an array of v16bf fragments indexed f = (t * nK32 + kt) * 32 + lane,
// where t = 16-column tile, kt = 32-deep K tile, lane = wave lane.
// Lane holds column n = t*16 + (lane&15); elems 0..7 -> k = c0+i, 8..15 -> c0+16+i,
// c0 = kt*32 + (lane>>4)*8.  K-tail zero-filled here so GEMM needs no guards.
__global__ void pack_w_kernel(const float* __restrict__ W, int K, int Nout,
                              __bf16* __restrict__ Wp)
{
    int nK32 = (K + 31) >> 5;
    int tid = blockIdx.x * blockDim.x + threadIdx.x;
    int total = (Nout >> 4) * nK32 * 32;
    if (tid >= total) return;
    int lane = tid & 31;
    int tmp  = tid >> 5;
    int kt   = tmp % nK32;
    int t    = tmp / nK32;
    int half = lane >> 4;
    int n    = t * 16 + (lane & 15);
    int c0   = kt * 32 + half * 8;
    __bf16* o = Wp + (size_t)tid * 16;
#pragma unroll
    for (int i = 0; i < 8; ++i) {
        int ka = c0 + i;
        int kb = c0 + 16 + i;
        o[i]     = (__bf16)((ka < K) ? W[(size_t)ka * Nout + n] : 0.f);
        o[i + 8] = (__bf16)((kb < K) ? W[(size_t)kb * Nout + n] : 0.f);
    }
}

// ---------------- WMMA GEMM: C[M x Nout] = act(A[M x K] @ W[K x Nout] + bias) --
// A row-major fp32, row stride lda (multiple of 32, zero-padded K..lda).
// Wp: pre-packed bf16 B fragments (see pack_w_kernel).
// One wave -> 16x64 output tile: 4 x v_wmma_f32_16x16x32_bf16 per K-step,
// sharing a single A fragment.  M%16==0, Nout%64==0 guaranteed by caller.
__global__ void __launch_bounds__(256)
wmma_gemm_kernel(const float* __restrict__ A, int lda,
                 const __bf16* __restrict__ Wp, const float* __restrict__ bias,
                 float* __restrict__ C, int ldc,
                 int M, int K, int Nout, int act)
{
    const int gtid   = blockIdx.x * blockDim.x + threadIdx.x;
    const int wave   = gtid >> 5;
    const int lane   = threadIdx.x & 31;
    const int tilesN = Nout >> 6;            // 64-col wave tiles
    const int tm = wave / tilesN;
    const int tn = wave % tilesN;
    if (tm * 16 >= M) return;                // whole-wave uniform exit

    const int half = lane >> 4;
    const int l15  = lane & 15;
    const int row  = tm * 16 + l15;
    const int nK32 = (K + 31) >> 5;

    const float* __restrict__ Arow = A + (size_t)row * lda;
    const v16bf* __restrict__ Wv = (const v16bf*)Wp;
    // fragment bases for the 4 consecutive 16-col tiles of this wave
    size_t f0 = ((size_t)(tn * 4 + 0) * nK32) * 32 + lane;
    size_t f1 = ((size_t)(tn * 4 + 1) * nK32) * 32 + lane;
    size_t f2 = ((size_t)(tn * 4 + 2) * nK32) * 32 + lane;
    size_t f3 = ((size_t)(tn * 4 + 3) * nK32) * 32 + lane;

    v8f acc0 = {}, acc1 = {}, acc2 = {}, acc3 = {};

    for (int kt = 0; kt < nK32; ++kt) {
        const int c0 = kt * 32 + half * 8;
        if (kt + 1 < nK32) __builtin_prefetch(Arow + c0 + 32, 0, 1);  // global_prefetch_b8

        // A fragment: fp32 -> bf16 on the fly (b128 loads, no guards: A zero-padded)
        float4 fa0 = *(const float4*)(Arow + c0);
        float4 fa1 = *(const float4*)(Arow + c0 + 4);
        float4 fa2 = *(const float4*)(Arow + c0 + 16);
        float4 fa3 = *(const float4*)(Arow + c0 + 20);
        v16bf a;
        a[0]  = (__bf16)fa0.x; a[1]  = (__bf16)fa0.y; a[2]  = (__bf16)fa0.z; a[3]  = (__bf16)fa0.w;
        a[4]  = (__bf16)fa1.x; a[5]  = (__bf16)fa1.y; a[6]  = (__bf16)fa1.z; a[7]  = (__bf16)fa1.w;
        a[8]  = (__bf16)fa2.x; a[9]  = (__bf16)fa2.y; a[10] = (__bf16)fa2.z; a[11] = (__bf16)fa2.w;
        a[12] = (__bf16)fa3.x; a[13] = (__bf16)fa3.y; a[14] = (__bf16)fa3.z; a[15] = (__bf16)fa3.w;

        // B fragments: single 32B vector load each (2x global_load_b128), pre-packed
        const size_t ko = (size_t)kt * 32;
        v16bf b0 = Wv[f0 + ko];
        v16bf b1 = Wv[f1 + ko];
        v16bf b2 = Wv[f2 + ko];
        v16bf b3 = Wv[f3 + ko];

        acc0 = __builtin_amdgcn_wmma_f32_16x16x32_bf16(false, a, false, b0, (short)0, acc0, false, false);
        acc1 = __builtin_amdgcn_wmma_f32_16x16x32_bf16(false, a, false, b1, (short)0, acc1, false, false);
        acc2 = __builtin_amdgcn_wmma_f32_16x16x32_bf16(false, a, false, b2, (short)0, acc2, false, false);
        acc3 = __builtin_amdgcn_wmma_f32_16x16x32_bf16(false, a, false, b3, (short)0, acc3, false, false);
    }

    // C/D layout: VGPR r holds row m = r + 8*half, col n = lane&15 of each 16-tile
    const int col0 = tn * 64 + l15;
    const float bz0 = bias ? bias[col0]      : 0.f;
    const float bz1 = bias ? bias[col0 + 16] : 0.f;
    const float bz2 = bias ? bias[col0 + 32] : 0.f;
    const float bz3 = bias ? bias[col0 + 48] : 0.f;
#pragma unroll
    for (int r = 0; r < 8; ++r) {
        int m = tm * 16 + r + 8 * half;
        float* Cm = C + (size_t)m * ldc + col0;
        Cm[0]  = act_apply(acc0[r] + bz0, act);
        Cm[16] = act_apply(acc1[r] + bz1, act);
        Cm[32] = act_apply(acc2[r] + bz2, act);
        Cm[48] = act_apply(acc3[r] + bz3, act);
    }
}

// ---------------- fills -------------------------------------------------------
__global__ void fill_f32(float* p, float v, int n) {
    int t = blockIdx.x * blockDim.x + threadIdx.x;
    if (t < n) p[t] = v;
}
__global__ void fill_u32(unsigned* p, unsigned v, int n) {
    int t = blockIdx.x * blockDim.x + threadIdx.x;
    if (t < n) p[t] = v;
}

// ---------------- geometry helpers --------------------------------------------
__device__ __forceinline__ void cross3(const float* a, const float* b, float* o) {
    o[0] = a[1] * b[2] - a[2] * b[1];
    o[1] = a[2] * b[0] - a[0] * b[2];
    o[2] = a[0] * b[1] - a[1] * b[0];
}
__device__ __forceinline__ void norm3(const float* a, float* o) {
    float n = sqrtf(a[0] * a[0] + a[1] * a[1] + a[2] * a[2]);
    n = fmaxf(n, 1e-12f);
    o[0] = a[0] / n; o[1] = a[1] / n; o[2] = a[2] / n;
}

// orientations: O[n] = rows [o1, n2, o1 x n2] from X[3n..3n+2]; last node -> 0
__global__ void orient_kernel(const float* __restrict__ pos, float* __restrict__ O) {
    int n = blockIdx.x * blockDim.x + threadIdx.x;
    if (n >= NN) return;
    float* on = O + n * 9;
    if (n == NN - 1) {
#pragma unroll
        for (int i = 0; i < 9; ++i) on[i] = 0.f;
        return;
    }
    const float* X0 = pos + n * 9;
    const float* X1 = X0 + 3;
    const float* X2 = X0 + 6;
    float d0[3] = {X1[0] - X0[0], X1[1] - X0[1], X1[2] - X0[2]};
    float d1[3] = {X2[0] - X1[0], X2[1] - X1[1], X2[2] - X1[2]};
    float u2[3], u1[3], n2[3], o1[3], t[3], o3[3];
    norm3(d0, u2);
    norm3(d1, u1);
    cross3(u2, u1, t);  norm3(t, n2);
    t[0] = u2[0] - u1[0]; t[1] = u2[1] - u1[1]; t[2] = u2[2] - u1[2];
    norm3(t, o1);
    cross3(o1, n2, o3);
    on[0] = o1[0]; on[1] = o1[1]; on[2] = o1[2];
    on[3] = n2[0]; on[4] = n2[1]; on[5] = n2[2];
    on[6] = o3[0]; on[7] = o3[1]; on[8] = o3[2];
}

// edge attributes: 46 features, stored with stride 64 (zero padded for WMMA K)
__global__ void edge_attr_kernel(const float* __restrict__ x,
                                 const float* __restrict__ pos,
                                 const int* __restrict__ src,
                                 const int* __restrict__ dst,
                                 const float* __restrict__ a_index,
                                 const float* __restrict__ order,
                                 const float* __restrict__ O,
                                 float* __restrict__ ea)
{
    int e = blockIdx.x * blockDim.x + threadIdx.x;
    if (e >= EE) return;
    int s = src[e], d = dst[e];
    float out[46];
    out[0] = a_index[s] - a_index[d];
#pragma unroll
    for (int c = 0; c < 20; ++c) out[1 + c] = x[s * 20 + c] - x[d * 20 + c];

    const float* cs = pos + s * 9;
    const float* cd = pos + d * 9;
    float rvec[9];
#pragma unroll
    for (int r = 0; r < 3; ++r) {
        float dx = cs[3 * r]     - cd[3 * r];
        float dy = cs[3 * r + 1] - cd[3 * r + 1];
        float dz = cs[3 * r + 2] - cd[3 * r + 2];
        float rij = sqrtf(dx * dx + dy * dy + dz * dz);
        out[21 + r] = expf(-rij);
        float inv = 1.f / fmaxf(rij, 1e-12f);
        rvec[3 * r]     = dx * inv;
        rvec[3 * r + 1] = dy * inv;
        rvec[3 * r + 2] = dz * inv;
    }
#pragma unroll
    for (int i = 0; i < 9; ++i) out[24 + i] = rvec[i];

    const float* Os = O + s * 9;
    const float* Od = O + d * 9;
#pragma unroll
    for (int i = 0; i < 3; ++i)
#pragma unroll
        for (int k = 0; k < 3; ++k) {
            float acc = 0.f;
#pragma unroll
            for (int j = 0; j < 3; ++j) acc += Os[3 * j + i] * Od[3 * j + k];
            out[33 + 3 * i + k] = acc;
        }
    float vm[3];
    norm3(rvec + 3, vm);
#pragma unroll
    for (int i = 0; i < 3; ++i) {
        float acc = 0.f;
#pragma unroll
        for (int j = 0; j < 3; ++j) acc += Os[3 * j + i] * vm[j];
        out[42 + i] = acc;
    }
    out[45] = order[e];

    float* row = ea + (long long)e * 64;
#pragma unroll
    for (int c = 0; c < 46; ++c) row[c] = out[c];
#pragma unroll
    for (int c = 46; c < 64; ++c) row[c] = 0.f;
}

// h0 = [x | pos] padded to stride 32
__global__ void h0_kernel(const float* __restrict__ x, const float* __restrict__ pos,
                          float* __restrict__ h) {
    int t = blockIdx.x * blockDim.x + threadIdx.x;
    if (t >= NN * 32) return;
    int n = t >> 5, c = t & 31;
    float v = 0.f;
    if (c < 20)      v = x[n * 20 + c];
    else if (c < 29) v = pos[n * 9 + (c - 20)];
    h[t] = v;
}

__global__ void count_kernel(const int* __restrict__ batch, float* __restrict__ counts) {
    int n = blockIdx.x * blockDim.x + threadIdx.x;
    if (n < NN) atomicAdd(&counts[batch[n]], 1.f);
}

// wave per edge: logits + running segment max
__global__ void __launch_bounds__(256)
attn_logits_kernel(const float* __restrict__ q, const float* __restrict__ k,
                   const float* __restrict__ e, int co,
                   const int* __restrict__ src, const int* __restrict__ dst,
                   float* __restrict__ logit, unsigned* __restrict__ mmax)
{
    int wave = (blockIdx.x * blockDim.x + threadIdx.x) >> 5;
    int lane = threadIdx.x & 31;
    if (wave >= EE) return;
    int s = src[wave], d = dst[wave];
    const float* qd = q + (long long)d * co;
    const float* ks = k + (long long)s * co;
    const float* ee = e + (long long)wave * co;
    float sum = 0.f;
    for (int c = lane; c < co; c += 32) sum += qd[c] * (ks[c] + ee[c]);
#pragma unroll
    for (int off = 16; off; off >>= 1) sum += __shfl_xor(sum, off, 32);
    if (lane == 0) {
        float lg = sum * rsqrtf((float)co);
        logit[wave] = lg;
        atomicMax(&mmax[d], fenc(lg));
    }
}

__global__ void alpha_kernel(const float* __restrict__ logit,
                             const unsigned* __restrict__ mmax,
                             const int* __restrict__ dst,
                             float* __restrict__ alpha, float* __restrict__ ssum)
{
    int e = blockIdx.x * blockDim.x + threadIdx.x;
    if (e >= EE) return;
    int d = dst[e];
    float a = expf(logit[e] - fdec(mmax[d]));
    alpha[e] = a;
    atomicAdd(&ssum[d], a);
}

// wave per edge: weighted scatter of (v[src] + e) into hout (pre-seeded with skip)
__global__ void __launch_bounds__(256)
agg_kernel(const float* __restrict__ v, const float* __restrict__ e,
           const float* __restrict__ alpha, const float* __restrict__ ssum,
           const int* __restrict__ src, const int* __restrict__ dst,
           int co, float* __restrict__ hout)
{
    int wave = (blockIdx.x * blockDim.x + threadIdx.x) >> 5;
    int lane = threadIdx.x & 31;
    if (wave >= EE) return;
    int s = src[wave], d = dst[wave];
    float w = alpha[wave] / fmaxf(ssum[d], 1e-16f);
    const float* vs = v + (long long)s * co;
    const float* ee = e + (long long)wave * co;
    float* hd = hout + (long long)d * co;
    for (int c = lane; c < co; c += 32) atomicAdd(&hd[c], (vs[c] + ee[c]) * w);
}

// per-node pooling head: sigmoid(f2(relu(f1(h)))) -> segment sum by batch
__global__ void pool_head_kernel(const float* __restrict__ h, int co,
                                 const float* __restrict__ W1, const float* __restrict__ b1,
                                 const float* __restrict__ W2, const float* __restrict__ b2,
                                 const int* __restrict__ batch, float* __restrict__ pooled)
{
    int n = blockIdx.x * blockDim.x + threadIdx.x;
    if (n >= NN) return;
    float hid[16];
#pragma unroll
    for (int j = 0; j < 16; ++j) hid[j] = b1[j];
    const float* hn = h + (long long)n * co;
    for (int c = 0; c < co; ++c) {
        float hv = hn[c];
#pragma unroll
        for (int j = 0; j < 16; ++j) hid[j] += hv * W1[c * 16 + j];
    }
#pragma unroll
    for (int j = 0; j < 16; ++j) hid[j] = fmaxf(hid[j], 0.f);
    int b = batch[n];
#pragma unroll
    for (int k = 0; k < 8; ++k) {
        float acc = b2[k];
#pragma unroll
        for (int j = 0; j < 16; ++j) acc += hid[j] * W2[j * 8 + k];
        float f = 1.f / (1.f + expf(-acc));
        atomicAdd(&pooled[b * 8 + k], f);
    }
}

// (pooled/counts) @ Wo + bo, accumulated into phsum   [grid=B, block=64]
__global__ void ph_o_kernel(const float* __restrict__ pooled, const float* __restrict__ counts,
                            const float* __restrict__ Wo, const float* __restrict__ bo,
                            float* __restrict__ phsum)
{
    int b = blockIdx.x, o = threadIdx.x;
    float cnt = fmaxf(counts[b], 1.f);
    float acc = bo[o];
#pragma unroll
    for (int k = 0; k < 8; ++k) acc += (pooled[b * 8 + k] / cnt) * Wo[k * 64 + o];
    phsum[b * 64 + o] += acc;
}

// small dense layer on few rows: out = act(scale * in @ W + b)  [grid=rows, block=Nout]
__global__ void dense_rows_kernel(const float* __restrict__ in, int ldin, float scale,
                                  const float* __restrict__ W, const float* __restrict__ b,
                                  float* __restrict__ out, int ldout, int K, int Nout, int act)
{
    int r = blockIdx.x, o = threadIdx.x;
    float acc = b ? b[o] : 0.f;
    for (int k = 0; k < K; ++k) acc += scale * in[r * ldin + k] * W[k * Nout + o];
    out[r * ldout + o] = act_apply(acc, act);
}

__global__ void relu_kernel(float* p, long long n) {
    long long t = (long long)blockIdx.x * blockDim.x + threadIdx.x;
    if (t < n) p[t] = fmaxf(p[t], 0.f);
}

// comb = [h(128) | ph_out[batch](64)]  stride 192
__global__ void comb_kernel(const float* __restrict__ h, const float* __restrict__ ph,
                            const int* __restrict__ batch, float* __restrict__ comb)
{
    long long t = (long long)blockIdx.x * blockDim.x + threadIdx.x;
    if (t >= (long long)NN * 192) return;
    int n = (int)(t / 192), c = (int)(t % 192);
    comb[t] = (c < 128) ? h[(long long)n * 128 + c] : ph[batch[n] * 64 + (c - 128)];
}

// final readout: out[n] = r1[n,:] @ W(128x1) + b
__global__ void readout2_kernel(const float* __restrict__ r1, const float* __restrict__ W,
                                const float* __restrict__ b, float* __restrict__ out)
{
    int n = blockIdx.x * blockDim.x + threadIdx.x;
    if (n >= NN) return;
    float acc = b[0];
    const float* rn = r1 + (long long)n * 128;
    for (int c = 0; c < 128; ++c) acc += rn[c] * W[c];
    out[n] = acc;
}

// =============================================================================
extern "C" void kernel_launch(void* const* d_in, const int* in_sizes, int n_in,
                              void* d_out, int out_size, void* d_ws, size_t ws_size,
                              hipStream_t stream)
{
    const float* x          = (const float*)d_in[0];
    const float* pos        = (const float*)d_in[1];
    const int*   edge_index = (const int*)d_in[2];
    const float* a_index    = (const float*)d_in[3];
    const float* order      = (const float*)d_in[4];
    const int*   batch      = (const int*)d_in[5];
    const int* src = edge_index;
    const int* dst = edge_index + EE;

    struct Lin { const float *W, *b; };
    int p = 6;
    auto nextWB = [&](bool bias) -> Lin {
        Lin l; l.W = (const float*)d_in[p++];
        l.b = bias ? (const float*)d_in[p++] : nullptr;
        return l;
    };
    Lin Lq[4], Lk[4], Lv[4], Le[4], Ls[4];
    for (int l = 0; l < 4; ++l) {
        Lq[l] = nextWB(true); Lk[l] = nextWB(true); Lv[l] = nextWB(true);
        Le[l] = nextWB(false); Ls[l] = nextWB(true);
    }
    Lin F1[4], F2[4], Lo[4];
    for (int l = 0; l < 4; ++l) { F1[l] = nextWB(true); F2[l] = nextWB(true); Lo[l] = nextWB(true); }
    Lin PP0 = nextWB(true), PP1 = nextWB(true);
    Lin PH0 = nextWB(true), PH1 = nextWB(true);
    Lin R0  = nextWB(true), R1  = nextWB(true);

    // ---- workspace carve-up (floats, 64B-aligned chunks) ----
    float* ws = (float*)d_ws;
    size_t off = 0;
    auto alloc = [&](size_t n) { float* r = ws + off; off += (n + 15) & ~(size_t)15; return r; };
    float*  O      = alloc((size_t)NN * 9);
    float*  ea     = alloc((size_t)EE * 64);
    float*  hA     = alloc((size_t)NN * 256);
    float*  hB     = alloc((size_t)NN * 256);
    float*  qb     = alloc((size_t)NN * 256);
    float*  kb     = alloc((size_t)NN * 256);
    float*  vb     = alloc((size_t)NN * 256);
    float*  ebuf   = alloc((size_t)EE * 256);
    float*  logit  = alloc(EE);
    float*  alpha  = alloc(EE);
    float*  mbuf   = alloc(NN);
    float*  ssum   = alloc(NN);
    float*  counts = alloc(NB);
    float*  pooled = alloc(NB * 8);
    float*  phsum  = alloc(NB * 64);
    float*  phhid  = alloc(NB * 128);
    float*  phout  = alloc(NB * 64);
    __bf16* Wp     = (__bf16*)alloc(32768);   // packed weights scratch (<=128KB)

    auto fillf = [&](float* ptr, float v, int n) {
        hipLaunchKernelGGL(fill_f32, dim3((n + 255) / 256), dim3(256), 0, stream, ptr, v, n);
    };
    auto fillu = [&](unsigned* ptr, unsigned v, int n) {
        hipLaunchKernelGGL(fill_u32, dim3((n + 255) / 256), dim3(256), 0, stream, ptr, v, n);
    };
    auto gemm = [&](const float* A, int lda, Lin L, float* Cm, int ldc,
                    int M, int K, int Nout, int act) {
        int nK32 = (K + 31) / 32;
        int ptot = (Nout / 16) * nK32 * 32;
        hipLaunchKernelGGL(pack_w_kernel, dim3((ptot + 255) / 256), dim3(256), 0, stream,
                           L.W, K, Nout, Wp);
        int waves  = (M / 16) * (Nout / 64);
        int blocks = (waves * 32 + 255) / 256;
        hipLaunchKernelGGL(wmma_gemm_kernel, dim3(blocks), dim3(256), 0, stream,
                           A, lda, Wp, L.b, Cm, ldc, M, K, Nout, act);
    };
    const int NBLK = (NN + 255) / 256;
    const int EBLK = (EE + 255) / 256;
    const int EWAVE_BLK = (EE * 32 + 255) / 256;

    // ---- preamble ----
    fillf(counts, 0.f, NB);
    hipLaunchKernelGGL(count_kernel, dim3(NBLK), dim3(256), 0, stream, batch, counts);
    hipLaunchKernelGGL(orient_kernel, dim3(NBLK), dim3(256), 0, stream, pos, O);
    hipLaunchKernelGGL(edge_attr_kernel, dim3(EBLK), dim3(256), 0, stream,
                       x, pos, src, dst, a_index, order, O, ea);
    hipLaunchKernelGGL(h0_kernel, dim3((NN * 32 + 255) / 256), dim3(256), 0, stream, x, pos, hA);
    fillf(phsum, 0.f, NB * 64);

    static const int CI[4] = {29, 128, 256, 128};
    static const int CO[4] = {128, 256, 128, 64};

    float* hin = hA;
    int ldin = 32;
    for (int l = 0; l < 4; ++l) {
        int ci = CI[l], co = CO[l];
        float* hout = (l & 1) ? hA : hB;

        gemm(hin, ldin, Lq[l], qb,   co, NN, ci, co, ACT_NONE);
        gemm(hin, ldin, Lk[l], kb,   co, NN, ci, co, ACT_NONE);
        gemm(hin, ldin, Lv[l], vb,   co, NN, ci, co, ACT_NONE);
        gemm(hin, ldin, Ls[l], hout, co, NN, ci, co, ACT_NONE);   // skip term seeds hout
        gemm(ea, 64,   Le[l], ebuf, co, EE, 46, co, ACT_NONE);    // edge linear (biggest GEMM)

        fillu((unsigned*)mbuf, 0u, NN);
        fillf(ssum, 0.f, NN);
        hipLaunchKernelGGL(attn_logits_kernel, dim3(EWAVE_BLK), dim3(256), 0, stream,
                           qb, kb, ebuf, co, src, dst, logit, (unsigned*)mbuf);
        hipLaunchKernelGGL(alpha_kernel, dim3(EBLK), dim3(256), 0, stream,
                           logit, (unsigned*)mbuf, dst, alpha, ssum);
        hipLaunchKernelGGL(agg_kernel, dim3(EWAVE_BLK), dim3(256), 0, stream,
                           vb, ebuf, alpha, ssum, src, dst, co, hout);

        fillf(pooled, 0.f, NB * 8);
        hipLaunchKernelGGL(pool_head_kernel, dim3(NBLK), dim3(256), 0, stream,
                           hout, co, F1[l].W, F1[l].b, F2[l].W, F2[l].b, batch, pooled);
        hipLaunchKernelGGL(ph_o_kernel, dim3(NB), dim3(64), 0, stream,
                           pooled, counts, Lo[l].W, Lo[l].b, phsum);

        long long tot = (long long)NN * co;
        hipLaunchKernelGGL(relu_kernel, dim3((unsigned)((tot + 255) / 256)), dim3(256), 0, stream,
                           hout, tot);
        hin = hout; ldin = co;
    }

    // ---- prepool: 64 -> 256 (leaky) -> 128 ----
    gemm(hin, 64, PP0, qb, 256, NN, 64, 256, ACT_LEAKY);
    gemm(qb, 256, PP1, hB, 128, NN, 256, 128, ACT_NONE);

    // ---- ph head: mean of 4 -> 64->128 leaky -> 128->64 ----
    hipLaunchKernelGGL(dense_rows_kernel, dim3(NB), dim3(128), 0, stream,
                       phsum, 64, 0.25f, PH0.W, PH0.b, phhid, 128, 64, 128, ACT_LEAKY);
    hipLaunchKernelGGL(dense_rows_kernel, dim3(NB), dim3(64), 0, stream,
                       phhid, 128, 1.0f, PH1.W, PH1.b, phout, 64, 128, 64, ACT_NONE);

    // ---- readout ----
    hipLaunchKernelGGL(comb_kernel, dim3((unsigned)(((long long)NN * 192 + 255) / 256)), dim3(256),
                       0, stream, hB, phout, batch, kb);
    gemm(kb, 192, R0, vb, 128, NN, 192, 128, ACT_LEAKY);
    hipLaunchKernelGGL(readout2_kernel, dim3(NBLK), dim3(256), 0, stream,
                       vb, R1.W, R1.b, (float*)d_out);
}
